// PMF_80161269612812
// MI455X (gfx1250) — matrix operand order repaired and checked by reading, not compiled, not run on hardware
//
#include <hip/hip_runtime.h>

typedef __attribute__((ext_vector_type(16))) _Float16 v16h;
typedef __attribute__((ext_vector_type(8)))  _Float16 v8h;
typedef __attribute__((ext_vector_type(8)))  float    v8f;

#define NUSERS 8192
#define NITEMS 8192
#define KDIM   256
#define LAMBDA 0.01f

#define TM   128   // block tile rows (users)
#define TN   128   // block tile cols (items)
#define KC   32    // K chunk == one wmma f16 k-step
#define LDSK 40    // padded LDS row stride in halves (32 + 8 pad -> bank-conflict-free)

union Frag16 { v16h v; v8h h[2]; };

// ---------------------------------------------------------------------------
// Kernel 1: split f32 -> (hi,lo) f16 pair, fused sum-of-squares (regularizer)
// ---------------------------------------------------------------------------
__global__ void __launch_bounds__(256)
pmf_split_f16(const float* __restrict__ X,
              _Float16* __restrict__ Xh,
              _Float16* __restrict__ Xl,
              float* __restrict__ regPart)
{
    __shared__ float red[8];
    const int t = threadIdx.x;
    const long base = ((long)blockIdx.x * 256 + t) * 8;
    const float4* X4 = (const float4*)(X + base);
    float4 f0 = X4[0];
    float4 f1 = X4[1];
    float xs[8] = {f0.x, f0.y, f0.z, f0.w, f1.x, f1.y, f1.z, f1.w};
    v8h hh, hl;
    float ss = 0.f;
#pragma unroll
    for (int i = 0; i < 8; ++i) {
        float x = xs[i];
        _Float16 h = (_Float16)x;
        _Float16 l = (_Float16)(x - (float)h);   // exact residual, fits f16 range
        hh[i] = h;
        hl[i] = l;
        ss = fmaf(x, x, ss);
    }
    *(v8h*)(Xh + base) = hh;
    *(v8h*)(Xl + base) = hl;

#pragma unroll
    for (int o = 16; o; o >>= 1) ss += __shfl_down(ss, o, 32);
    if ((t & 31) == 0) red[t >> 5] = ss;
    __syncthreads();
    if (t == 0) {
        float s = 0.f;
#pragma unroll
        for (int w = 0; w < 8; ++w) s += red[w];
        regPart[blockIdx.x] = s;
    }
}

// ---------------------------------------------------------------------------
// Kernel 2: per-tile masked squared-error via split-f16 WMMA (3 wmma / k-step)
// ---------------------------------------------------------------------------
__global__ void __launch_bounds__(256)
pmf_loss_tile(const float* __restrict__ R, const float* __restrict__ I,
              const _Float16* __restrict__ Uh, const _Float16* __restrict__ Ul,
              const _Float16* __restrict__ Vh, const _Float16* __restrict__ Vl,
              float* __restrict__ mainPart)
{
    __shared__ _Float16 sUh[TM * LDSK];
    __shared__ _Float16 sUl[TM * LDSK];
    __shared__ _Float16 sVh[TN * LDSK];
    __shared__ _Float16 sVl[TN * LDSK];
    __shared__ float red[8];

    const int t    = threadIdx.x;
    const int lane = t & 31;
    const int wave = t >> 5;          // 8 waves
    const int lm   = lane & 15;
    const int sel  = lane >> 4;       // 0/1 half-wave selector (ISA frag layout)
    const int wm   = wave >> 2;       // 0..1 -> 64-row band
    const int wn   = wave & 3;        // 0..3 -> 32-col band
    const int tm0  = blockIdx.y * TM;
    const int tn0  = blockIdx.x * TN;

    // Prefetch the streaming R/I tile; it is consumed only in the epilogue,
    // so these global_prefetch_b8 overlap the entire WMMA main loop.
    {
        const long gn = tn0 + wn * 32 + lm;
#pragma unroll
        for (int mi = 0; mi < 4; ++mi) {
            const long gm = tm0 + wm * 64 + mi * 16 + sel * 8;
#pragma unroll
            for (int j = 0; j < 8; ++j) {
                __builtin_prefetch(R + (gm + j) * (long)NITEMS + gn, 0, 0);
                __builtin_prefetch(I + (gm + j) * (long)NITEMS + gn, 0, 0);
            }
        }
    }

    v8f acc[4][2] = {};

    for (int kc = 0; kc < KDIM / KC; ++kc) {
        __syncthreads();
        // Stage 128x32-half tiles of Uh/Ul/Vh/Vl into LDS (16 B per op, coalesced).
#pragma unroll
        for (int i = 0; i < 2; ++i) {
            const int gg  = t + i * 256;       // 512 groups of 8 halves per tile
            const int row = gg >> 2;
            const int col = (gg & 3) * 8;
            const long srcU = (long)(tm0 + row) * KDIM + kc * KC + col;
            const long srcV = (long)(tn0 + row) * KDIM + kc * KC + col;
            const int  dst  = row * LDSK + col;
            *(v8h*)&sUh[dst] = *(const v8h*)(Uh + srcU);
            *(v8h*)&sUl[dst] = *(const v8h*)(Ul + srcU);
            *(v8h*)&sVh[dst] = *(const v8h*)(Vh + srcV);
            *(v8h*)&sVl[dst] = *(const v8h*)(Vl + srcV);
        }
        __syncthreads();

        // A fragments: 16x32 f16, lane layout per ISA (VGPR0-3: K=sel*8.., VGPR4-7: K=16+sel*8..)
        Frag16 ah[4], al[4];
#pragma unroll
        for (int mi = 0; mi < 4; ++mi) {
            const int row = (wm * 64 + mi * 16 + lm) * LDSK;
            ah[mi].h[0] = *(const v8h*)&sUh[row + sel * 8];
            ah[mi].h[1] = *(const v8h*)&sUh[row + 16 + sel * 8];
            al[mi].h[0] = *(const v8h*)&sUl[row + sel * 8];
            al[mi].h[1] = *(const v8h*)&sUl[row + 16 + sel * 8];
        }
        // B fragments: 32x16, lane = column n, halves K=sel*8.. and K=16+sel*8.. of V's row n
#pragma unroll
        for (int ni = 0; ni < 2; ++ni) {
            const int col = (wn * 32 + ni * 16 + lm) * LDSK;
            Frag16 bh, bl;
            bh.h[0] = *(const v8h*)&sVh[col + sel * 8];
            bh.h[1] = *(const v8h*)&sVh[col + 16 + sel * 8];
            bl.h[0] = *(const v8h*)&sVl[col + sel * 8];
            bl.h[1] = *(const v8h*)&sVl[col + 16 + sel * 8];
#pragma unroll
            for (int mi = 0; mi < 4; ++mi) {
                acc[mi][ni] = __builtin_amdgcn_wmma_f32_16x16x32_f16(
                    false, ah[mi].v, false, bh.v, (short)0, acc[mi][ni], false, false);
                acc[mi][ni] = __builtin_amdgcn_wmma_f32_16x16x32_f16(
                    false, ah[mi].v, false, bl.v, (short)0, acc[mi][ni], false, false);
                acc[mi][ni] = __builtin_amdgcn_wmma_f32_16x16x32_f16(
                    false, al[mi].v, false, bh.v, (short)0, acc[mi][ni], false, false);
            }
        }
    }

    // Epilogue: stream R/I in the C-fragment layout, accumulate I*(R-pred)^2.
    float part = 0.f;
#pragma unroll
    for (int mi = 0; mi < 4; ++mi) {
        const long gmBase = tm0 + wm * 64 + mi * 16 + sel * 8;
#pragma unroll
        for (int ni = 0; ni < 2; ++ni) {
            const long gn = tn0 + wn * 32 + ni * 16 + lm;
#pragma unroll
            for (int j = 0; j < 8; ++j) {
                const long off = (gmBase + j) * (long)NITEMS + gn;
                const float r = R[off];
                const float w = I[off];
                const float e = r - acc[mi][ni][j];
                part = fmaf(w * e, e, part);
            }
        }
    }
#pragma unroll
    for (int o = 16; o; o >>= 1) part += __shfl_down(part, o, 32);
    if (lane == 0) red[wave] = part;
    __syncthreads();
    if (t == 0) {
        float s = 0.f;
#pragma unroll
        for (int w = 0; w < 8; ++w) s += red[w];
        mainPart[blockIdx.y * gridDim.x + blockIdx.x] = s;
    }
}

// ---------------------------------------------------------------------------
// Kernel 3: deterministic fixed-order final reduction (no float atomics)
// ---------------------------------------------------------------------------
__global__ void __launch_bounds__(256)
pmf_reduce(const float* __restrict__ mainPart, int nMain,
           const float* __restrict__ regPart, int nReg,
           float* __restrict__ out)
{
    __shared__ float red[8];
    const int t = threadIdx.x;
    float s = 0.f;
    for (int i = t; i < nMain; i += 256) s += mainPart[i];
    float r = 0.f;
    for (int i = t; i < nReg; i += 256) r += regPart[i];
    float v = fmaf(LAMBDA, r, s);
#pragma unroll
    for (int o = 16; o; o >>= 1) v += __shfl_down(v, o, 32);
    if ((t & 31) == 0) red[t >> 5] = v;
    __syncthreads();
    if (t == 0) {
        float tot = 0.f;
#pragma unroll
        for (int w = 0; w < 8; ++w) tot += red[w];
        out[0] = tot;
    }
}

// ---------------------------------------------------------------------------
extern "C" void kernel_launch(void* const* d_in, const int* in_sizes, int n_in,
                              void* d_out, int out_size, void* d_ws, size_t ws_size,
                              hipStream_t stream)
{
    const float* R = (const float*)d_in[0];
    const float* I = (const float*)d_in[1];
    const float* U = (const float*)d_in[2];
    const float* V = (const float*)d_in[3];

    char* ws = (char*)d_ws;
    const size_t szUV = (size_t)NUSERS * KDIM * 2;   // 4 MiB per f16 matrix
    _Float16* Uh = (_Float16*)(ws);
    _Float16* Ul = (_Float16*)(ws + szUV);
    _Float16* Vh = (_Float16*)(ws + 2 * szUV);
    _Float16* Vl = (_Float16*)(ws + 3 * szUV);
    float* mainPart = (float*)(ws + 4 * szUV);
    const int nMain = (NUSERS / TM) * (NITEMS / TN);       // 4096
    float* regPart = mainPart + nMain;
    const int nRegU = (NUSERS * KDIM) / (256 * 8);         // 1024
    const int nRegV = (NITEMS * KDIM) / (256 * 8);         // 1024

    pmf_split_f16<<<nRegU, 256, 0, stream>>>(U, Uh, Ul, regPart);
    pmf_split_f16<<<nRegV, 256, 0, stream>>>(V, Vh, Vl, regPart + nRegU);

    dim3 grid(NITEMS / TN, NUSERS / TM);
    pmf_loss_tile<<<grid, 256, 0, stream>>>(R, I, Uh, Ul, Vh, Vl, mainPart);

    pmf_reduce<<<1, 256, 0, stream>>>(mainPart, nMain, regPart, nRegU + nRegV,
                                      (float*)d_out);
}
